// VMDPeriodDetection_12421045420581
// MI455X (gfx1250) — matrix operand order repaired and checked by reading, not compiled, not run on hardware
//
#include <hip/hip_runtime.h>
#include <hip/hip_bf16.h>
#include <math.h>

typedef _Float16 v16h __attribute__((ext_vector_type(16)));
typedef float    v8f  __attribute__((ext_vector_type(8)));

#define T_LEN  8192
#define F_LEN  4097   // T/2 + 1
#define NCH    64
#define NB     64
#define KMODES 5

// ---------------------------------------------------------------------------
// Kernel 1: channel_var[b] = mean_c var_t(x[b,:,c], ddof=1)
// Bandwidth-bound streaming pass (134 MB total). float4 loads: 512 B per
// wave-instruction (global_load_b128), 16 lanes cover one 64-channel row.
// ---------------------------------------------------------------------------
__global__ void __launch_bounds__(1024)
vmd_var_kernel(const float* __restrict__ x, float* __restrict__ var_out) {
  const int b   = blockIdx.x;
  const int tid = threadIdx.x;
  const int cg  = tid & 15;                    // channel group: channels 4*cg..4*cg+3
  const int t0  = tid >> 4;                    // 0..63 time slice
  const float4* xb = (const float4*)(x + (size_t)b * T_LEN * NCH) + cg;

  float s1x = 0.f, s1y = 0.f, s1z = 0.f, s1w = 0.f;
  float s2x = 0.f, s2y = 0.f, s2z = 0.f, s2w = 0.f;
  for (int t = t0; t < T_LEN; t += 64) {
    float4 v = xb[(size_t)t * 16];
    s1x += v.x; s2x += v.x * v.x;
    s1y += v.y; s2y += v.y * v.y;
    s1z += v.z; s2z += v.z * v.z;
    s1w += v.w; s2w += v.w * v.w;
    if (t + 256 < T_LEN)
      __builtin_prefetch(&xb[(size_t)(t + 256) * 16], 0, 0);  // 4 iters ahead
  }

  __shared__ float sh1[64 * 64];               // [slice][channel]
  __shared__ float sh2[64 * 64];
  {
    const int base = t0 * 64 + cg * 4;
    sh1[base + 0] = s1x; sh2[base + 0] = s2x;
    sh1[base + 1] = s1y; sh2[base + 1] = s2y;
    sh1[base + 2] = s1z; sh2[base + 2] = s2z;
    sh1[base + 3] = s1w; sh2[base + 3] = s2w;
  }
  __syncthreads();
  for (int s = 32; s > 0; s >>= 1) {           // reduce 64 slices -> slice 0
    for (int i = tid; i < s * 64; i += 1024) {
      int sl = i >> 6, c = i & 63;
      sh1[sl * 64 + c] += sh1[(sl + s) * 64 + c];
      sh2[sl * 64 + c] += sh2[(sl + s) * 64 + c];
    }
    __syncthreads();
  }
  if (tid < 64) {                              // per-channel ddof=1 variance
    float S1 = sh1[tid], S2 = sh2[tid];
    sh1[tid] = (S2 - S1 * S1 * (1.0f / T_LEN)) * (1.0f / (T_LEN - 1));
  }
  __syncthreads();
  for (int s = 32; s > 0; s >>= 1) {           // sum 64 channel variances
    if (tid < s) sh1[tid] += sh1[tid + s];
    __syncthreads();
  }
  if (tid == 0) var_out[b] = sh1[0] * (1.0f / NCH);
}

// ---------------------------------------------------------------------------
// Kernel 2: rep = argmax over 64 batch variances (ties -> lowest index)
// ---------------------------------------------------------------------------
__global__ void vmd_rep_kernel(const float* __restrict__ var_in, int* __restrict__ rep_out) {
  if (threadIdx.x == 0) {
    float best = var_in[0]; int bi = 0;
    for (int i = 1; i < NB; ++i) { float v = var_in[i]; if (v > best) { best = v; bi = i; } }
    rep_out[0] = bi;
  }
}

// ---------------------------------------------------------------------------
// Kernel 3: Cooley-Tukey stage A as WMMA GEMM.
//   t = n1 + 64*n2, n1 in [0,64), n2 in [0,128)
//   G[r,n1] = sum_n2 x[n1+64*n2] * (cos(2*pi*r*n2/128) - i*sin(...))
// => two real GEMMs [128x128] x [128x64] on v_wmma_f32_16x16x32_f16.
// One wave per 16x16 output tile; 8 mtiles x 4 ntiles = 32 waves.
// ---------------------------------------------------------------------------
__global__ void __launch_bounds__(256)
vmd_stageA_wmma_kernel(const float* __restrict__ x, const int* __restrict__ rep,
                       float* __restrict__ Gc, float* __restrict__ Gs) {
  const int lane   = threadIdx.x & 31;
  const int w      = blockIdx.x * (blockDim.x >> 5) + (threadIdx.x >> 5);
  const int mtile  = w >> 2;        // 0..7  (r tile)
  const int ntile  = w & 3;         // 0..3  (n1 tile)
  const int laneHi = lane >> 4;
  const int lmod   = lane & 15;
  const float* xb = x + (size_t)rep[0] * (size_t)(T_LEN * NCH);

  v8f accC = {};
  v8f accS = {};
  const float w128 = 6.28318530717958647692f / 128.0f;
#pragma unroll
  for (int step = 0; step < 4; ++step) {        // K = 128 in chunks of 32
    v16h ac, as, bf;
#pragma unroll
    for (int e = 0; e < 16; ++e) {
      // A fragment (16x32 f16): K = e + 16*laneHi
      int ka = 32 * step + e + 16 * laneHi;
      int r  = mtile * 16 + lmod;
      float th = w128 * (float)((r * ka) & 127);
      ac[e] = (_Float16)__cosf(th);
      as[e] = (_Float16)(-__sinf(th));
      // B fragment (32x16 f16): K = e + 16*laneHi, N = lane%16
      int kb = 32 * step + e + 16 * laneHi;
      int n1 = ntile * 16 + lmod;
      bf[e] = (_Float16)xb[(size_t)(n1 + 64 * kb) * NCH];   // channel 0
    }
    accC = __builtin_amdgcn_wmma_f32_16x16x32_f16(false, ac, false, bf, (short)0, accC, false, false);
    accS = __builtin_amdgcn_wmma_f32_16x16x32_f16(false, as, false, bf, (short)0, accS, false, false);
  }
#pragma unroll
  for (int v = 0; v < 8; ++v) {                 // C/D: M = v + 8*laneHi, N = lane%16
    int r  = mtile * 16 + v + 8 * laneHi;
    int n1 = ntile * 16 + lmod;
    Gc[r * 64 + n1] = accC[v];
    Gs[r * 64 + n1] = accS[v];
  }
}

// ---------------------------------------------------------------------------
// Kernel 4: stage B — twiddle recombination.
//   X[k] = sum_n1 (cos(phi) - i sin(phi)) * (Gc + i Gs),  phi = 2*pi*k*n1/8192
// One thread per frequency bin; 64-step complex rotation recurrence.
// ---------------------------------------------------------------------------
__global__ void __launch_bounds__(256)
vmd_stageB_kernel(const float* __restrict__ Gc, const float* __restrict__ Gs,
                  float* __restrict__ sre, float* __restrict__ sim) {
  int k = blockIdx.x * blockDim.x + threadIdx.x;
  if (k >= F_LEN) return;
  int r = k & 127;
  float s1, c1;
  __sincosf(6.28318530717958647692f * (float)k / (float)T_LEN, &s1, &c1);
  float cw = 1.f, sw = 0.f, Xr = 0.f, Xi = 0.f;
  const float* gc = Gc + r * 64;
  const float* gs = Gs + r * 64;
#pragma unroll 8
  for (int n1 = 0; n1 < 64; ++n1) {
    float a = gc[n1], b = gs[n1];
    Xr += cw * a + sw * b;
    Xi += cw * b - sw * a;
    float ncw = cw * c1 - sw * s1;
    sw = sw * c1 + cw * s1;
    cw = ncw;
  }
  sre[k] = Xr; sim[k] = Xi;
}

// ---------------------------------------------------------------------------
// Kernel 5: full ADMM VMD (50 iters) with the entire state held in registers.
// Each of 1024 threads owns 4-5 frequency columns x 5 modes. Then the spectrum
// argmax (spec == |u_hat|, DC killed, Re-only at Nyquist), periods, top-k,
// and output write. Only omega / convergence need cross-thread reduction.
// ---------------------------------------------------------------------------
__global__ void __launch_bounds__(1024)
vmd_iterate_kernel(const float* __restrict__ sig_re, const float* __restrict__ sig_im,
                   const int* __restrict__ kin, float* __restrict__ out) {
  const int tid  = threadIdx.x;
  const int lane = tid & 31;
  const int wid  = tid >> 5;

  float sr[5], si[5], lr[5], li[5], fr[5];
  float ur[5][KMODES], ui[5][KMODES];
  int nc = 0;
#pragma unroll
  for (int j = 0; j < 5; ++j) {
    int f = tid + j * 1024;
    if (f < F_LEN) {
      sr[j] = sig_re[f]; si[j] = sig_im[f];
      lr[j] = 0.f; li[j] = 0.f;
      fr[j] = (float)f * (1.0f / T_LEN);
#pragma unroll
      for (int k = 0; k < KMODES; ++k) { ur[j][k] = 0.f; ui[j][k] = 0.f; }
      nc = j + 1;
    }
  }

  __shared__ float omg[KMODES];
  __shared__ int   doneFlag;
  __shared__ float red[32 * 11];
  if (tid < KMODES) omg[tid] = 0.f;
  if (tid == 0) doneFlag = 0;
  __syncthreads();

  for (int it = 0; it < 50; ++it) {
    float vals[11];
#pragma unroll
    for (int j = 0; j < 11; ++j) vals[j] = 0.f;
    float om[KMODES];
#pragma unroll
    for (int k = 0; k < KMODES; ++k) om[k] = omg[k];

#pragma unroll
    for (int j = 0; j < 5; ++j) {
      if (j < nc) {
        float sumr = 0.f, sumi = 0.f;
#pragma unroll
        for (int k = 0; k < KMODES; ++k) { sumr += ur[j][k]; sumi += ui[j][k]; }
        float baser = sr[j] - sumr + 0.5f * lr[j];
        float basei = si[j] - sumi + 0.5f * li[j];
        float nsr = 0.f, nsi = 0.f;
#pragma unroll
        for (int k = 0; k < KMODES; ++k) {
          float numr = baser + ur[j][k];
          float numi = basei + ui[j][k];
          float d   = fr[j] - om[k];
          float inv = 1.0f / (1.0f + 1000.0f * d * d);
          float unr = numr * inv, uni = numi * inv;
          float pw  = unr * unr + uni * uni;
          vals[k]     += fr[j] * pw;
          vals[5 + k] += pw;
          float dr = unr - ur[j][k], di = uni - ui[j][k];
          vals[10] += dr * dr + di * di;
          ur[j][k] = unr; ui[j][k] = uni;
          nsr += unr; nsi += uni;
        }
        lr[j] += 1e-6f * (sr[j] - nsr);
        li[j] += 1e-6f * (si[j] - nsi);
      }
    }
    // reduce 11 scalars: wave32 shuffle then LDS
#pragma unroll
    for (int off = 16; off > 0; off >>= 1)
#pragma unroll
      for (int j = 0; j < 11; ++j) vals[j] += __shfl_down(vals[j], off);
    if (lane == 0)
#pragma unroll
      for (int j = 0; j < 11; ++j) red[wid * 11 + j] = vals[j];
    __syncthreads();
    if (tid == 0) {
      float tot[11];
#pragma unroll
      for (int j = 0; j < 11; ++j) tot[j] = 0.f;
      for (int w = 0; w < 32; ++w)
#pragma unroll
        for (int j = 0; j < 11; ++j) tot[j] += red[w * 11 + j];
#pragma unroll
      for (int k = 0; k < KMODES; ++k) omg[k] = tot[k] / (tot[5 + k] + 1e-10f);
      float conv = tot[10] * (1.0f / ((float)KMODES * (float)F_LEN));
      doneFlag = (it > 0 && conv < 1e-7f) ? 1 : 0;
    }
    __syncthreads();
    if (doneFlag) break;
  }

  // ---- spectrum argmax per mode (spec == |u_hat|; DC killed, Re at Nyquist)
  float bP[KMODES]; int bF[KMODES];
#pragma unroll
  for (int k = 0; k < KMODES; ++k) { bP[k] = -1.f; bF[k] = F_LEN; }
#pragma unroll
  for (int j = 0; j < 5; ++j) {
    if (j < nc) {
      int f = tid + j * 1024;
#pragma unroll
      for (int k = 0; k < KMODES; ++k) {
        float p;
        if (f == 0)                p = 0.f;
        else if (f == T_LEN / 2)   p = ur[j][k] * ur[j][k];
        else                       p = ur[j][k] * ur[j][k] + ui[j][k] * ui[j][k];
        if (p > bP[k]) { bP[k] = p; bF[k] = f; }
      }
    }
  }
#pragma unroll
  for (int off = 16; off > 0; off >>= 1) {
#pragma unroll
    for (int k = 0; k < KMODES; ++k) {
      float op = __shfl_down(bP[k], off);
      int   of = __shfl_down(bF[k], off);
      if (op > bP[k] || (op == bP[k] && of < bF[k])) { bP[k] = op; bF[k] = of; }
    }
  }
  __shared__ float sP[32 * KMODES];
  __shared__ int   sF[32 * KMODES];
  if (lane == 0) {
#pragma unroll
    for (int k = 0; k < KMODES; ++k) { sP[wid * KMODES + k] = bP[k]; sF[wid * KMODES + k] = bF[k]; }
  }
  __syncthreads();
  if (tid == 0) {
    float gP[KMODES]; int gF[KMODES];
#pragma unroll
    for (int k = 0; k < KMODES; ++k) { gP[k] = -1.f; gF[k] = F_LEN; }
    for (int w = 0; w < 32; ++w)
#pragma unroll
      for (int k = 0; k < KMODES; ++k) {
        float p = sP[w * KMODES + k]; int f = sF[w * KMODES + k];
        if (p > gP[k] || (p == gP[k] && f < gF[k])) { gP[k] = p; gF[k] = f; }
      }
    int periods[KMODES]; float energy[KMODES];
#pragma unroll
    for (int k = 0; k < KMODES; ++k) {
      int idx = gF[k] < 1 ? 1 : gF[k];
      periods[k] = T_LEN / idx;
      energy[k]  = sqrtf(gP[k] > 0.f ? gP[k] : 0.f);
    }
    int kk = kin[0]; if (kk > KMODES) kk = KMODES; if (kk < 1) kk = 1;
    // stable descending selection (matches lax.top_k tie behavior)
    int order[KMODES]; bool used[KMODES] = {false, false, false, false, false};
    for (int j = 0; j < KMODES; ++j) {
      int bi = 0; float bv = -1e30f;
      for (int m = 0; m < KMODES; ++m)
        if (!used[m] && energy[m] > bv) { bv = energy[m]; bi = m; }
      used[bi] = true; order[j] = bi;
    }
    for (int j = 0; j < kk; ++j) out[j] = (float)periods[order[j]];
    for (int b = 0; b < NB; ++b)
      for (int j = 0; j < kk; ++j) out[kk + b * kk + j] = energy[order[j]];
  }
}

// ---------------------------------------------------------------------------
extern "C" void kernel_launch(void* const* d_in, const int* in_sizes, int n_in,
                              void* d_out, int out_size, void* d_ws, size_t ws_size,
                              hipStream_t stream) {
  const float* x  = (const float*)d_in[0];
  const int* kin  = (const int*)d_in[1];
  float* W   = (float*)d_ws;
  float* var = W;                    // 64
  int*   rep = (int*)(W + 64);       // 1 (padded to 128)
  float* Gc  = W + 128;              // 128*64
  float* Gs  = Gc + 128 * 64;        // 128*64
  float* sre = Gs + 128 * 64;        // 4097 (padded to 4112)
  float* sim = sre + 4112;           // 4097
  float* out = (float*)d_out;

  vmd_var_kernel<<<NB, 1024, 0, stream>>>(x, var);
  vmd_rep_kernel<<<1, 64, 0, stream>>>(var, rep);
  vmd_stageA_wmma_kernel<<<4, 256, 0, stream>>>(x, rep, Gc, Gs);
  vmd_stageB_kernel<<<17, 256, 0, stream>>>(Gc, Gs, sre, sim);
  vmd_iterate_kernel<<<1, 1024, 0, stream>>>(sre, sim, kin, out);
}
